// DynamicConv_55808805044522
// MI455X (gfx1250) — matrix operand order repaired and checked by reading, not compiled, not run on hardware
//
#include <hip/hip_runtime.h>
#include <stdint.h>

// Problem constants (from reference): B=32, C=128, O=128, H=W=64, HID=64, K=3
#define BB  32
#define CC  128
#define OO  128
#define HH  64
#define WWD 64
#define HID 64
#define NS  9      // K*K
#define LDS_W 66   // 64 cols + 1-col halo each side

typedef __bf16 bf16;
typedef __attribute__((ext_vector_type(16))) bf16  v16bf;
typedef __attribute__((ext_vector_type(8)))  bf16  v8bf;
typedef __attribute__((ext_vector_type(8)))  float v8f;

static __device__ __forceinline__ unsigned short f2bf_bits(float f) {
    union { float f; unsigned u; } cv; cv.f = f;
    unsigned u = cv.u;
    unsigned r = u + 0x7FFFu + ((u >> 16) & 1u);   // round-to-nearest-even
    return (unsigned short)(r >> 16);
}

// ---------------- Stage 1: global average pool: pooled[b][c] ----------------
__global__ __launch_bounds__(256) void pool_kernel(const float* __restrict__ x,
                                                   float* __restrict__ pooled) {
    const int c = blockIdx.x, b = blockIdx.y;
    const float* p = x + ((size_t)(b * CC + c)) * (HH * WWD);
    float s = 0.f;
    for (int i = threadIdx.x; i < HH * WWD; i += 256) s += p[i];
    __shared__ float red[256];
    red[threadIdx.x] = s;
    __syncthreads();
    for (int off = 128; off > 0; off >>= 1) {
        if (threadIdx.x < off) red[threadIdx.x] += red[threadIdx.x + off];
        __syncthreads();
    }
    if (threadIdx.x == 0) pooled[b * CC + c] = red[0] * (1.0f / (HH * WWD));
}

// ---------------- Stage 2: h = relu(pooled @ W1^T + b1) ----------------
__global__ __launch_bounds__(256) void mlp_kernel(const float* __restrict__ pooled,
                                                  const float* __restrict__ W1,
                                                  const float* __restrict__ b1,
                                                  float* __restrict__ h) {
    const int idx = blockIdx.x * 256 + threadIdx.x;   // 0..2047 = b*HID + j
    const int b = idx / HID, j = idx % HID;
    if (b >= BB) return;
    float acc = b1[j];
    const float* pr = pooled + b * CC;
    const float* wr = W1 + j * CC;
    #pragma unroll 8
    for (int c = 0; c < CC; ++c) acc += pr[c] * wr[c];
    h[idx] = acc > 0.f ? acc : 0.f;
}

// ---- Stage 3: per-sample conv weights, bf16, layout wgen[b][s][o][c] ----
// w[b, r] = sum_j h[b,j]*W2[r,j] + b2[r],  r = (o*CC + c)*9 + s
__global__ __launch_bounds__(256) void wgen_kernel(const float* __restrict__ h,
                                                   const float* __restrict__ W2,
                                                   const float* __restrict__ b2,
                                                   unsigned short* __restrict__ wgen) {
    __shared__ float hs[BB * HID];   // 8 KB: all 32 hidden vectors
    for (int i = threadIdx.x; i < BB * HID; i += 256) hs[i] = h[i];
    __syncthreads();

    const int oc = blockIdx.x * 256 + threadIdx.x;   // one (o,c) per thread
    const int o = oc / CC, c = oc % CC;

    for (int s = 0; s < NS; ++s) {
        const size_t r = (size_t)(o * CC + c) * NS + s;
        const float* wrow = W2 + r * HID;
        float rowreg[HID];
        #pragma unroll
        for (int j = 0; j < HID; ++j) rowreg[j] = wrow[j];
        const float bias = b2[r];
        #pragma unroll 1
        for (int b = 0; b < BB; ++b) {
            const float* hb = hs + b * HID;
            float acc = bias;
            #pragma unroll
            for (int j = 0; j < HID; ++j) acc += rowreg[j] * hb[j];
            wgen[(((size_t)b * NS + s) * OO + o) * CC + c] = f2bf_bits(acc);
        }
    }
}

// ---------------- Stage 4: conv as 9 shifted GEMMs on WMMA ----------------
// Block: one sample, all O=128, 2 output rows (N=128 px). 8 waves = 4(M)x2(row).
// LDS tile: [4 input rows][66 cols][32 ch] bf16, pixel-major/channel-contig so a
// B-fragment (16 consecutive K=channels for one pixel) is one 32B contiguous load.
__global__ __launch_bounds__(256) void dynconv_kernel(const float* __restrict__ x,
                                                      const unsigned short* __restrict__ wgen,
                                                      float* __restrict__ out) {
    __shared__ unsigned short tile[4 * LDS_W * 32];   // 16.5 KB

    const int b    = blockIdx.y;
    const int h0   = blockIdx.x * 2;          // first output row of this block
    const int tid  = threadIdx.x;
    const int wv   = tid >> 5;
    const int lane = tid & 31;
    const int half = lane >> 4;               // 0/1 (K-half selector)
    const int ln   = lane & 15;               // row within M-tile / col within N-tile
    const int obase = (wv >> 1) * 32;         // wave M-base: 0,32,64,96
    const int rr    = wv & 1;                 // wave output row within pair

    v8f acc[2][4];
    #pragma unroll
    for (int i = 0; i < 2; ++i)
        #pragma unroll
        for (int j = 0; j < 4; ++j)
            acc[i][j] = (v8f){0.f, 0.f, 0.f, 0.f, 0.f, 0.f, 0.f, 0.f};

    const float* xb = x + (size_t)b * CC * HH * WWD;
    const unsigned short* wb = wgen + (size_t)b * NS * OO * CC;

    for (int c0 = 0; c0 < CC; c0 += 32) {     // channel chunks (WMMA K=32)
        __syncthreads();
        // Stage 32 channels x 4 rows x 66 cols with transpose + fp32->bf16 + pad.
        // 8448 elems = 256 threads * 33 iters exactly (no divergence).
        for (int e = tid; e < 4 * LDS_W * 32; e += 256) {
            const int wcol = e % LDS_W;
            const int t    = e / LDS_W;
            const int ch   = t & 31;
            const int r    = t >> 5;
            const int gr = h0 - 1 + r;        // input row (with halo)
            const int gw = wcol - 1;          // input col (with halo)
            float v = 0.f;
            if ((unsigned)gr < (unsigned)HH && (unsigned)gw < (unsigned)WWD)
                v = xb[((size_t)(c0 + ch) * HH + gr) * WWD + gw];
            tile[(r * LDS_W + wcol) * 32 + ch] = f2bf_bits(v);
        }
        __syncthreads();

        #pragma unroll
        for (int kh = 0; kh < 3; ++kh) {
            const int lr = rr + kh;           // LDS row index 0..3
            #pragma unroll
            for (int kw = 0; kw < 3; ++kw) {
                const int s = kh * 3 + kw;
                const unsigned short* as = wb + (size_t)s * OO * CC;
                // A fragments: lane(half,m) holds K run {16h..16h+15}
                v16bf a[2];
                #pragma unroll
                for (int mt = 0; mt < 2; ++mt) {
                    const unsigned short* arow = as + (size_t)(obase + mt * 16 + ln) * CC + c0;
                    v8bf lo = *(const v8bf*)(arow + 16 * half);
                    v8bf hi = *(const v8bf*)(arow + 16 * half + 8);
                    a[mt] = __builtin_shufflevector(lo, hi,
                        0, 1, 2, 3, 4, 5, 6, 7, 8, 9, 10, 11, 12, 13, 14, 15);
                }
                // Preload ALL 4 B fragments into distinct registers so the
                // scheduler can clause the 8 ds_load_b128s, wait once, then
                // issue the 8 WMMAs back-to-back (no per-pair dscnt drains).
                v16bf bfr[4];
                #pragma unroll
                for (int j = 0; j < 4; ++j) {
                    const unsigned short* bp =
                        &tile[((lr * LDS_W) + j * 16 + ln + kw) * 32 + 16 * half];
                    bfr[j] = *(const v16bf*)bp;   // 32B aligned contiguous
                }
                #pragma unroll
                for (int j = 0; j < 4; ++j) {
                    acc[0][j] = __builtin_amdgcn_wmma_f32_16x16x32_bf16(
                        false, a[0], false, bfr[j], (short)0, acc[0][j], false, false);
                    acc[1][j] = __builtin_amdgcn_wmma_f32_16x16x32_bf16(
                        false, a[1], false, bfr[j], (short)0, acc[1][j], false, false);
                }
            }
        }
    }

    // Epilogue: C/D layout -> VGPR v is row (v + 8*half), col = ln
    float* outp = out + (size_t)b * OO * HH * WWD;
    const int orow = h0 + rr;
    #pragma unroll
    for (int mt = 0; mt < 2; ++mt)
        #pragma unroll
        for (int j = 0; j < 4; ++j)
            #pragma unroll
            for (int v = 0; v < 8; ++v) {
                const int m = obase + mt * 16 + v + 8 * half;
                outp[(size_t)m * HH * WWD + (size_t)orow * WWD + j * 16 + ln] = acc[mt][j][v];
            }
}

extern "C" void kernel_launch(void* const* d_in, const int* in_sizes, int n_in,
                              void* d_out, int out_size, void* d_ws, size_t ws_size,
                              hipStream_t stream) {
    (void)in_sizes; (void)n_in; (void)out_size; (void)ws_size;
    const float* x  = (const float*)d_in[0];
    const float* W1 = (const float*)d_in[1];
    const float* b1 = (const float*)d_in[2];
    const float* W2 = (const float*)d_in[3];
    const float* b2 = (const float*)d_in[4];
    float* out = (float*)d_out;

    // Workspace layout: wgen bf16 [32][9][128][128] = 9,437,184 B,
    // then pooled f32 [32][128] (16 KB), then h f32 [32][64] (8 KB).
    unsigned short* wgen = (unsigned short*)d_ws;
    float* pooled = (float*)((char*)d_ws + 9437184);
    float* h      = (float*)((char*)d_ws + 9437184 + 16384);

    pool_kernel<<<dim3(CC, BB), 256, 0, stream>>>(x, pooled);
    mlp_kernel<<<(BB * HID) / 256, 256, 0, stream>>>(pooled, W1, b1, h);
    wgen_kernel<<<(OO * CC) / 256, 256, 0, stream>>>(h, W2, b2, wgen);
    dynconv_kernel<<<dim3(HH / 2, BB), 256, 0, stream>>>(x, wgen, out);
}
